// DurationFusionModule_23072564314586
// MI455X (gfx1250) — compile-verified, matrix operand order checked by reading, not run on hardware
//
#include <hip/hip_runtime.h>

// ---------------- CDNA5 WMMA / vector types ----------------
typedef __attribute__((ext_vector_type(16))) __bf16 v16bf;
typedef __attribute__((ext_vector_type(8)))  float  v8f;
typedef __attribute__((ext_vector_type(4)))  float  v4f;
typedef __attribute__((ext_vector_type(4)))  __bf16 v4bf;
typedef __attribute__((ext_vector_type(2)))  __bf16 v2bf;

#define AS1 __attribute__((address_space(1)))
typedef const float AS1 * gcf;
typedef const v4f   AS1 * gcv4f;
typedef const int   AS1 * gci;

#define NTHREADS 512
#define NWAVES   16
#define TB       16     // batches per block
#define NBLOCKS  128    // 2048 / TB
#define QSTR     1032   // padded [batch][col] stride for Qchunk / Wepi

// LDS layout (bytes) — total 325184 (159 x 2KB blocks) <= 320KB WGP LDS
#define OFF_D1T   0          // [16][256] bf16            8192
#define OFF_D2T   8192       // [16][256] bf16            8192
#define OFF_WSTG  16384      // [256 c][34] bf16 pairs   17408
#define OFF_QCH   33792      // [16 b][1032] bf16        33024
#define OFF_WEPI  66816      // [16 b][1032] bf16        33024 (persistent w = d2@Ww+bw)
#define OFF_WKT   99840      // [64 d][64 dd] bf16        8192
#define OFF_WVT   108032     // [64 d][64 dd] bf16        8192
#define OFF_WAVE  116224     // 16 x 13056              208896
#define OFF_RED   325120     // [16] f32                    64
#define LDS_TOTAL 325184
// per-wave region: r3bf 4096 | kbuf 4096 | scor 4096 | stats 768  = 13056

// element j (0..15) of a v16bf A/B fragment -> K index (ISA 7.12.2)
__device__ __forceinline__ int wmma_k(int j, int half) {
    int i = j >> 1;
    return ((i & 3) << 1) + ((i >> 2) << 4) + (half << 3) + (j & 1);
}
__device__ __forceinline__ v8f wmma_bf16(v16bf a, v16bf b, v8f c) {
    return __builtin_amdgcn_wmma_f32_16x16x32_bf16(false, a, false, b, (short)0, c, false, false);
}
// force global (addrspace 1) loads
__device__ __forceinline__ v4f   ldg4(const float* p) { return *(gcv4f)(gcf)p; }
__device__ __forceinline__ float ldg (const float* p) { return *(gcf)p; }
__device__ __forceinline__ int   ldgi(const int*   p) { return *(gci)p; }

// Cooperative GEMM:  out[batch][col] = sum_e W[e][col] * Bsrc[batch][e] + bias[col]
// 64 column-tiles (1024 cols): global col = colBase + (ct>>2)*strideS + (ct&3)*16 + m
// wave w owns ct = st*16 + w;  K = 256 in 8 WMMA steps.
__device__ void gemmT(const float* W, int ldw, const float* bias,
                      int colBase, int strideS,
                      const __bf16* Bsrc,   // [16][256] bf16 (D1^T or D2^T)
                      __bf16* Wstage,       // [256 c][34] bf16, k-pairs contiguous
                      __bf16* outLDS,       // [16 b][QSTR] bf16
                      int tid, int wave, int lane)
{
    const int half = lane >> 4, lm = lane & 15;
#pragma unroll 1
    for (int st = 0; st < 4; ++st) {
        const int gct = st * 16 + wave;
        v8f acc;
#pragma unroll
        for (int q = 0; q < 8; ++q) acc[q] = 0.f;

#pragma unroll 1
        for (int ks = 0; ks < 8; ++ks) {
            __syncthreads();   // previous Wstage consumers done
            v4f t0[2], t1[2];
#pragma unroll
            for (int rr = 0; rr < 2; ++rr) {   // stage [32 e][256 c], e-pairs
                int u  = rr * NTHREADS + tid;
                int ep = u >> 6;               // 0..15 e-pair
                int c  = (u & 63) * 4;
                int ct = st * 16 + (c >> 4);
                int col = colBase + (ct >> 2) * strideS + ((ct & 3) << 4) + (c & 15);
                const float* p = W + (ks * 32 + 2 * ep) * ldw + col;
                t0[rr] = ldg4(p);
                t1[rr] = ldg4(p + ldw);
            }
            if (ks < 7)
                __builtin_prefetch(W + (ks + 1) * 32 * ldw + colBase, 0, 1);
#pragma unroll
            for (int rr = 0; rr < 2; ++rr) {
                int u  = rr * NTHREADS + tid;
                int ep = u >> 6;
                int c  = (u & 63) * 4;
#pragma unroll
                for (int i = 0; i < 4; ++i) {
                    v2bf pk; pk[0] = (__bf16)t0[rr][i]; pk[1] = (__bf16)t1[rr][i];
                    *(v2bf*)(Wstage + (c + i) * 34 + 2 * ep) = pk;
                }
            }
            __syncthreads();

            v16bf a, b;  // A[m=col][k=e], B[n=batch][k=e] — pair loads
#pragma unroll
            for (int jp = 0; jp < 8; ++jp) {
                int kk = wmma_k(2 * jp, half);
                v2bf pa = *(const v2bf*)(Wstage + (wave * 16 + lm) * 34 + kk);
                v2bf pv = *(const v2bf*)(Bsrc + lm * 256 + ks * 32 + kk);
                a[2*jp] = pa[0]; a[2*jp+1] = pa[1];
                b[2*jp] = pv[0]; b[2*jp+1] = pv[1];
            }
            acc = wmma_bf16(a, b, acc);
        }
        // output (+bias): lane n = batch, paired column stores
        const int colb = colBase + (gct >> 2) * strideS + ((gct & 3) << 4);
#pragma unroll
        for (int rp = 0; rp < 4; ++rp) {
            int m0 = 2 * rp + half * 8;
            v2bf pk;
            pk[0] = (__bf16)(acc[2*rp]     + ldg(bias + colb + m0));
            pk[1] = (__bf16)(acc[2*rp + 1] + ldg(bias + colb + m0 + 1));
            *(v2bf*)(outLDS + lm * QSTR + gct * 16 + m0) = pk;
        }
    }
    __syncthreads();
}

__global__ void __launch_bounds__(NTHREADS)
duration_fusion_kernel(const float* __restrict__ r,    const int* __restrict__ dIdx,
                       const float* __restrict__ emb1, const float* __restrict__ emb2,
                       const float* __restrict__ Wq,   const float* __restrict__ bq,
                       const float* __restrict__ Wk,   const float* __restrict__ bk,
                       const float* __restrict__ Wv,   const float* __restrict__ bv,
                       const float* __restrict__ Wb,   const float* __restrict__ bb,
                       const float* __restrict__ Ww,   const float* __restrict__ bw,
                       float* __restrict__ ws)
{
    extern __shared__ char smem[];
    __bf16* sD1T  = (__bf16*)(smem + OFF_D1T);
    __bf16* sD2T  = (__bf16*)(smem + OFF_D2T);
    __bf16* sWst  = (__bf16*)(smem + OFF_WSTG);
    __bf16* sQch  = (__bf16*)(smem + OFF_QCH);
    __bf16* sWepi = (__bf16*)(smem + OFF_WEPI);
    __bf16* sWkT  = (__bf16*)(smem + OFF_WKT);
    __bf16* sWvT  = (__bf16*)(smem + OFF_WVT);
    float*  sRed  = (float*)(smem + OFF_RED);

    const int tid  = threadIdx.x;
    const int wave = tid >> 5;
    const int lane = tid & 31;
    const int half = lane >> 4, lm = lane & 15;
    const int b0   = blockIdx.x * TB;
    const int myB  = b0 + wave;

    char*   wvb  = smem + OFF_WAVE + wave * 13056;
    __bf16* r3bf = (__bf16*)wvb;             // [32 t][64 dd]
    __bf16* kbuf = (__bf16*)(wvb + 4096);    // [32 t][64 d]
    float*  scor = (float*)(wvb + 8192);     // [32 s][32 t]
    float*  rowm = (float*)(wvb + 12288);    // [32]
    float*  rowd = rowm + 32;                // [32]
    float*  pb   = rowd + 32;                // [32]
    float*  gbuf = pb + 32;                  // [64]

    // ---- Phase 0: stage gathered embedding rows + r3 (vectorized, global AS) ----
#pragma unroll 1
    for (int li = tid * 4; li < TB * 256; li += NTHREADS * 4) {
        int j = li >> 8, e = li & 255;
        int dj = ldgi(dIdx + b0 + j);
        v4f x1 = ldg4(emb1 + (size_t)dj * 256 + e);
        v4f x2 = ldg4(emb2 + (size_t)dj * 256 + e);
        v4bf y1, y2;
#pragma unroll
        for (int q = 0; q < 4; ++q) { y1[q] = (__bf16)x1[q]; y2[q] = (__bf16)x2[q]; }
        *(v4bf*)(sD1T + j * 256 + e) = y1;
        *(v4bf*)(sD2T + j * 256 + e) = y2;
    }
#pragma unroll 1
    for (int i = lane * 4; i < 32 * 64; i += 128) {
        v4f x = ldg4(r + (size_t)myB * 2048 + i);
        v4bf y;
#pragma unroll
        for (int q = 0; q < 4; ++q) y[q] = (__bf16)x[q];
        *(v4bf*)(r3bf + i) = y;
    }
    __syncthreads();

    // ---- epilogue weights: w = d2 @ Ww + bw  (same WMMA routine as Q) ----
    gemmT(Ww, 1024, bw, 0, 64, sD2T, sWst, sQch, tid, wave, lane);
    // keep a persistent copy (sQch gets reused by the Q GEMMs)
#pragma unroll 1
    for (int i = tid; i < (TB * QSTR) / 2; i += NTHREADS)
        ((unsigned int*)sWepi)[i] = ((const unsigned int*)sQch)[i];

    // per-batch scalar: bscal = d2 @ Wb + bb
    {
        float p = 0.f;
        for (int e = lane; e < 256; e += 32)
            p += (float)sD2T[wave * 256 + e] * ldg(Wb + e);
        for (int off = 16; off; off >>= 1) p += __shfl_down(p, off, 32);
        if (lane == 0) ws[NBLOCKS + myB] = p + ldg(bb);
    }
    __syncthreads();

    float accSum = 0.f;

    // ---------------- head loop ----------------
#pragma unroll 1
    for (int h = 0; h < 16; ++h) {
        __syncthreads();   // protect previous head's sWkT/sWvT readers
#pragma unroll
        for (int rr = 0; rr < 2; ++rr) {   // stage Wk_h^T / Wv_h^T [64 d][64 dd]
            int li = rr * 2048 + tid * 4;
            int dd = li >> 6, dc = li & 63;
            v4f xk = ldg4(Wk + dd * 1024 + h * 64 + dc);
            v4f xv = ldg4(Wv + dd * 1024 + h * 64 + dc);
#pragma unroll
            for (int q = 0; q < 4; ++q) {
                sWkT[(dc + q) * 64 + dd] = (__bf16)xk[q];
                sWvT[(dc + q) * 64 + dd] = (__bf16)xv[q];
            }
        }
        __syncthreads();

        // reload r3 A-fragments from LDS (pair loads)
        v16bf rA00, rA01, rA10, rA11;
#pragma unroll
        for (int jp = 0; jp < 8; ++jp) {
            int kk = wmma_k(2 * jp, half);
            v2bf p0 = *(const v2bf*)(r3bf + lm * 64 + kk);
            v2bf p1 = *(const v2bf*)(r3bf + lm * 64 + 32 + kk);
            v2bf p2 = *(const v2bf*)(r3bf + (16 + lm) * 64 + kk);
            v2bf p3 = *(const v2bf*)(r3bf + (16 + lm) * 64 + 32 + kk);
            rA00[2*jp] = p0[0]; rA00[2*jp+1] = p0[1];
            rA01[2*jp] = p1[0]; rA01[2*jp+1] = p1[1];
            rA10[2*jp] = p2[0]; rA10[2*jp+1] = p2[1];
            rA11[2*jp] = p3[0]; rA11[2*jp+1] = p3[1];
        }

        // per-wave K GEMM: kbuf[t][d] = r3 @ Wk_h + bk
#pragma unroll 1
        for (int nt = 0; nt < 4; ++nt) {
            v16bf bk0, bk1;     // B[n=d][k=dd]
#pragma unroll
            for (int jp = 0; jp < 8; ++jp) {
                int kk = wmma_k(2 * jp, half);
                v2bf p0 = *(const v2bf*)(sWkT + (nt * 16 + lm) * 64 + kk);
                v2bf p1 = *(const v2bf*)(sWkT + (nt * 16 + lm) * 64 + 32 + kk);
                bk0[2*jp] = p0[0]; bk0[2*jp+1] = p0[1];
                bk1[2*jp] = p1[0]; bk1[2*jp+1] = p1[1];
            }
            v8f a0, a1;
#pragma unroll
            for (int q = 0; q < 8; ++q) { a0[q] = 0.f; a1[q] = 0.f; }
            a0 = wmma_bf16(rA00, bk0, a0);  a0 = wmma_bf16(rA01, bk1, a0);
            a1 = wmma_bf16(rA10, bk0, a1);  a1 = wmma_bf16(rA11, bk1, a1);
            const int dc = nt * 16 + lm;
            const float bkv = ldg(bk + h * 64 + dc);
#pragma unroll
            for (int q = 0; q < 8; ++q) {
                kbuf[(q + half * 8) * 64 + dc]      = (__bf16)(a0[q] + bkv);
                kbuf[(16 + q + half * 8) * 64 + dc] = (__bf16)(a1[q] + bkv);
            }
        }

        // Q (cooperative) + scores (per-wave), two s-halves
#pragma unroll 1
        for (int sh = 0; sh < 2; ++sh) {
            gemmT(Wq, 32768, bq, sh * 16384 + h * 64, 1024, sD1T, sWst, sQch,
                  tid, wave, lane);
            v16bf aq0, aq1;   // A[m=s][k=d] for my batch row (pair loads)
#pragma unroll
            for (int jp = 0; jp < 8; ++jp) {
                int kk = wmma_k(2 * jp, half);
                v2bf p0 = *(const v2bf*)(sQch + wave * QSTR + lm * 64 + kk);
                v2bf p1 = *(const v2bf*)(sQch + wave * QSTR + lm * 64 + 32 + kk);
                aq0[2*jp] = p0[0]; aq0[2*jp+1] = p0[1];
                aq1[2*jp] = p1[0]; aq1[2*jp+1] = p1[1];
            }
#pragma unroll 1
            for (int tt = 0; tt < 2; ++tt) {
                v16bf bt0, bt1;  // B[n=t][k=d] from kbuf (pair loads)
#pragma unroll
                for (int jp = 0; jp < 8; ++jp) {
                    int kk = wmma_k(2 * jp, half);
                    v2bf p0 = *(const v2bf*)(kbuf + (tt * 16 + lm) * 64 + kk);
                    v2bf p1 = *(const v2bf*)(kbuf + (tt * 16 + lm) * 64 + 32 + kk);
                    bt0[2*jp] = p0[0]; bt0[2*jp+1] = p0[1];
                    bt1[2*jp] = p1[0]; bt1[2*jp+1] = p1[1];
                }
                v8f a;
#pragma unroll
                for (int q = 0; q < 8; ++q) a[q] = 0.f;
                a = wmma_bf16(aq0, bt0, a);
                a = wmma_bf16(aq1, bt1, a);
#pragma unroll
                for (int q = 0; q < 8; ++q) {
                    int s = sh * 16 + q + half * 8;
                    int t = tt * 16 + lm;
                    scor[s * 32 + t] = a[q] * 0.125f;   // 1/sqrt(64)
                }
            }
        }

        // softmax row stats (lane = s)
        {
            float mx = -1e30f;
            for (int t = 0; t < 32; ++t) mx = fmaxf(mx, scor[lane * 32 + t]);
            float den = 0.f;
            for (int t = 0; t < 32; ++t) den += __expf(scor[lane * 32 + t] - mx);
            rowm[lane] = mx; rowd[lane] = den;
        }
        // pbar[t] = sum_s P[s][t]   (lane = t)
        {
            float p = 0.f;
            for (int s = 0; s < 32; ++s)
                p += __expf(scor[s * 32 + lane] - rowm[s]) / rowd[s];
            pb[lane] = p;
        }
        // g[dd] = pbar @ r3
#pragma unroll
        for (int dp = 0; dp < 2; ++dp) {
            int dd = lane + dp * 32;
            float g = 0.f;
            for (int t = 0; t < 32; ++t) g += pb[t] * (float)r3bf[t * 64 + dd];
            gbuf[dd] = g;
        }
        // out_h[d] = g @ Wv_h + 32*bv ; accumulate leaky(out)*w
#pragma unroll
        for (int dp = 0; dp < 2; ++dp) {
            int dc = lane + dp * 32;
            float o = 0.f;
            for (int dd = 0; dd < 64; ++dd)
                o += gbuf[dd] * (float)sWvT[dc * 64 + dd];
            o += 32.f * ldg(bv + h * 64 + dc);
            float lr = o > 0.f ? o : 0.01f * o;
            accSum += lr * (float)sWepi[wave * QSTR + h * 64 + dc];
        }
    }

    // ---- block reduction ----
    for (int off = 16; off; off >>= 1) accSum += __shfl_down(accSum, off, 32);
    if (lane == 0) sRed[wave] = accSum;
    __syncthreads();
    if (tid == 0) {
        float s = 0.f;
#pragma unroll
        for (int i = 0; i < NWAVES; ++i) s += sRed[i];
        ws[blockIdx.x] = s;
    }
}

// out[b] = sum(block partials) + bscal[b]   (deterministic reduction)
__global__ void finalize_kernel(const float* __restrict__ ws, float* __restrict__ out)
{
    int i = blockIdx.x * blockDim.x + threadIdx.x;
    if (i >= 2048) return;
    float t = 0.f;
    for (int k = 0; k < NBLOCKS; ++k) t += ws[k];
    out[i] = t + ws[NBLOCKS + i];
}

extern "C" void kernel_launch(void* const* d_in, const int* in_sizes, int n_in,
                              void* d_out, int out_size, void* d_ws, size_t ws_size,
                              hipStream_t stream) {
    const float* r    = (const float*)d_in[0];
    const int*   dIdx = (const int*)  d_in[1];
    const float* emb1 = (const float*)d_in[2];
    const float* emb2 = (const float*)d_in[3];
    const float* Wq   = (const float*)d_in[4];
    const float* bq   = (const float*)d_in[5];
    const float* Wk   = (const float*)d_in[6];
    const float* bk   = (const float*)d_in[7];
    const float* Wv   = (const float*)d_in[8];
    const float* bv   = (const float*)d_in[9];
    const float* Wb   = (const float*)d_in[10];
    const float* bb   = (const float*)d_in[11];
    const float* Ww   = (const float*)d_in[12];
    const float* bw   = (const float*)d_in[13];
    float* ws  = (float*)d_ws;
    float* out = (float*)d_out;

    duration_fusion_kernel<<<NBLOCKS, NTHREADS, LDS_TOTAL, stream>>>(
        r, dIdx, emb1, emb2, Wq, bq, Wk, bk, Wv, bv, Wb, bb, Ww, bw, ws);
    finalize_kernel<<<8, 256, 0, stream>>>(ws, out);
}
